// SuperSweetNet_20246475833999
// MI455X (gfx1250) — matrix operand order repaired and checked by reading, not compiled, not run on hardware
//
#include <hip/hip_runtime.h>
#include <cstddef>

typedef __attribute__((ext_vector_type(16))) _Float16 v16h;
typedef __attribute__((ext_vector_type(8)))  float    v8f;
typedef unsigned int u32x4 __attribute__((ext_vector_type(4)));
typedef int          i32x4 __attribute__((ext_vector_type(4)));
typedef int          i32x8 __attribute__((ext_vector_type(8)));

#define G_TOT 512
#define NPER  128
#define EPER  1024
#define HD    128
#define F_IN  133
#define K1 103
#define K2 83
#define K3 67
#define NEG 0.01f
#define EPSBN 1e-5f

// ---------------------------------------------------------------------------
// WMMA layouts (wave32, v_wmma_f32_16x16x32_f16):
//  A (16x32): lane<16: elems0..7 = K k0..k0+7, elems8..15 = K k0+16..23 (M=lane)
//             lane>=16: elems0..7 = K k0+8..15, elems8..15 = K k0+24..31
//  B (32x16): lane<16 holds col N=lane, K=k0..k0+7,k0+16..23 ; lane>=16: K=k0+8..15,k0+24..31
//  C/D: vgpr v, lane<16 -> (M=v, N=lane); lane>=16 -> (M=8+v, N=lane-16)
// B matrices are PRE-PACKED to f16 in fragment order:
//   Bp[(((ntile*nchunks)+kk)*32 + lane)*16 + e]  -> one 32B vector load / frag
// ---------------------------------------------------------------------------

// ================= Pack kernel: W[K x N] f32 -> WMMA-B f16 fragments =======
__global__ __launch_bounds__(256) void pack_b_kernel(const float* __restrict__ W,
                                                     _Float16* __restrict__ Bp,
                                                     int K, int N, int nchunks)
{
    const int idx = blockIdx.x * 256 + threadIdx.x;
    const int total = (N >> 4) * nchunks * 512;
    if (idx >= total) return;
    const int e    = idx & 15;
    const int lane = (idx >> 4) & 31;
    const int tile = idx >> 9;
    const int kk   = tile % nchunks;
    const int nt   = tile / nchunks;
    const int half = lane >> 4, l16 = lane & 15;
    const int k   = (kk << 5) + half * 8 + e + ((e >> 3) << 3);
    const int col = (nt << 4) + l16;
    Bp[idx] = (k < K) ? (_Float16)W[k * N + col] : (_Float16)0.0f;
}

// ============================ Kernel 1: Hlin = X @ W1 ======================
__global__ __launch_bounds__(256) void gemm1_kernel(const float* __restrict__ X,
                                                    const _Float16* __restrict__ Bp1,
                                                    float* __restrict__ Hlin)
{
    const int tid  = threadIdx.x;
    const int wave = tid >> 5, lane = tid & 31;
    const int half = lane >> 4, l16 = lane & 15;
    const size_t row0 = (size_t)blockIdx.x * 128 + (size_t)wave * 16;

    // A: K=133 padded to 160 -> 5 chunks of 32 (stride 133 => scalar loads)
    v16h A[5];
    const size_t arow = (row0 + (size_t)l16) * F_IN;
#pragma unroll
    for (int kk = 0; kk < 5; ++kk) {
        const int klo = kk * 32 + half * 8;
        const int khi = klo + 16;
#pragma unroll
        for (int e = 0; e < 8; ++e) {
            const int ka = klo + e, kb = khi + e;
            float v0 = (ka < F_IN) ? X[arow + ka] : 0.0f;
            float v1 = (kb < F_IN) ? X[arow + kb] : 0.0f;
            A[kk][e]     = (_Float16)v0;
            A[kk][8 + e] = (_Float16)v1;
        }
    }

#pragma unroll
    for (int nt = 0; nt < 8; ++nt) {
        v8f acc = {};
#pragma unroll
        for (int kk = 0; kk < 5; ++kk) {
            const v16h b = *(const v16h*)(Bp1 + (((nt * 5 + kk) * 32 + lane) << 4));
            acc = __builtin_amdgcn_wmma_f32_16x16x32_f16(false, A[kk], false, b,
                                                         (short)0, acc, false, false);
        }
        const int crow = half * 8;
        const int ccol = nt * 16 + l16;
#pragma unroll
        for (int v = 0; v < 8; ++v)
            Hlin[(row0 + crow + v) * HD + ccol] = acc[v];
    }
}

// ============================ Per-graph LDS arena ==========================
struct GLds {
    float h0[NPER * HD];     // 64 KB  (MUST be at offset 0: TDM lds_addr = 0)
    float h1[NPER * HD];     // 64 KB
    int   esrc[EPER];
    int   edst[EPER];
    int   eord[EPER];
    int   rowstart[NPER + 1];
    int   fillcnt[NPER];
    float deg[NPER];
    float invd[NPER];
    float score[NPER];
    int   sidx[NPER];
    int   remap[NPER];
    float pbuf[HD];
    float pinv;
    float zacc[2 * HD];
};

__device__ inline void gcn_agg(GLds& L, const float* __restrict__ hin,
                               float* __restrict__ hout,
                               const float* __restrict__ bias, int n, int tid)
{
    for (int i = tid; i < n; i += 256) { L.deg[i] = 1.0f; L.fillcnt[i] = 0; }
    __syncthreads();
    for (int e = tid; e < EPER; e += 256) {
        if (L.esrc[e] >= 0) {
            atomicAdd(&L.deg[L.edst[e]], 1.0f);
            atomicAdd(&L.fillcnt[L.edst[e]], 1);
        }
    }
    __syncthreads();
    for (int i = tid; i < n; i += 256) L.invd[i] = rsqrtf(L.deg[i]);
    if (tid == 0) {
        int acc = 0;
        for (int i = 0; i < n; ++i) { L.rowstart[i] = acc; acc += L.fillcnt[i]; }
        L.rowstart[n] = acc;
    }
    __syncthreads();
    for (int i = tid; i < n; i += 256) L.fillcnt[i] = 0;
    __syncthreads();
    for (int e = tid; e < EPER; e += 256) {
        int s = L.esrc[e];
        if (s >= 0) {
            int d = L.edst[e];
            int pos = L.rowstart[d] + atomicAdd(&L.fillcnt[d], 1);
            L.eord[pos] = e;
        }
    }
    __syncthreads();
    const int f = tid & (HD - 1);
    const int sub = tid >> 7;            // two dst nodes in flight
    for (int base = 0; base < n; base += 2) {
        int node = base + sub;
        if (node < n) {
            float acc = 0.0f;
            int jb = L.rowstart[node], je = L.rowstart[node + 1];
            for (int j = jb; j < je; ++j) {
                int s = L.esrc[L.eord[j]];
                acc += hin[s * HD + f] * L.invd[s];
            }
            float id = L.invd[node];
            float v = acc * id + hin[node * HD + f] * (id * id) + bias[f];
            hout[node * HD + f] = v > 0.0f ? v : NEG * v;
        }
    }
    __syncthreads();
}

__device__ inline void topk_pool(GLds& L, const float* __restrict__ hin,
                                 float* __restrict__ hout,
                                 const float* __restrict__ p, int n, int k, int tid)
{
    if (tid < HD) L.pbuf[tid] = p[tid];
    __syncthreads();
    if (tid == 0) {
        float s = 0.0f;
        for (int i = 0; i < HD; ++i) s += L.pbuf[i] * L.pbuf[i];
        L.pinv = rsqrtf(s);
    }
    __syncthreads();
    if (tid < NPER) {
        float s;
        if (tid < n) {
            s = 0.0f;
            for (int ff = 0; ff < HD; ++ff) {
                int fx = (ff + tid) & (HD - 1);      // skew: avoid bank conflicts
                s += hin[tid * HD + fx] * L.pbuf[fx];
            }
            s *= L.pinv;
        } else {
            s = -1e30f;
        }
        L.score[tid] = s;
        L.sidx[tid] = tid;
    }
    __syncthreads();
    for (int ksz = 2; ksz <= NPER; ksz <<= 1) {       // bitonic, descending
        for (int j = ksz >> 1; j > 0; j >>= 1) {
            if (tid < NPER) {
                int ixj = tid ^ j;
                if (ixj > tid) {
                    bool desc = ((tid & ksz) == 0);
                    float a = L.score[tid], b = L.score[ixj];
                    bool sw = desc ? (a < b) : (a > b);
                    if (sw) {
                        L.score[tid] = b; L.score[ixj] = a;
                        int t = L.sidx[tid]; L.sidx[tid] = L.sidx[ixj]; L.sidx[ixj] = t;
                    }
                }
            }
            __syncthreads();
        }
    }
    if (tid < NPER) L.remap[tid] = -1;
    __syncthreads();
    if (tid < k) L.remap[L.sidx[tid]] = tid;
    __syncthreads();
    const int f = tid & (HD - 1);
    const int sub = tid >> 7;
    for (int base = 0; base < k; base += 2) {
        int j = base + sub;
        if (j < k) {
            float gate = tanhf(L.score[j]);
            hout[j * HD + f] = hin[L.sidx[j] * HD + f] * gate;
        }
    }
    __syncthreads();
    for (int e = tid; e < EPER; e += 256) {
        int s = L.esrc[e];
        if (s >= 0) {
            int s2 = L.remap[s], d2 = L.remap[L.edst[e]];
            if (s2 >= 0 && d2 >= 0) { L.esrc[e] = s2; L.edst[e] = d2; }
            else L.esrc[e] = -1;
        }
    }
    __syncthreads();
}

__device__ inline void readout_add(GLds& L, const float* __restrict__ h, int k, int tid)
{
    __syncthreads();
    const int f = tid & (HD - 1);
    if (tid < HD) {
        float mx = -1e30f;
        for (int j = 0; j < k; ++j) mx = fmaxf(mx, h[j * HD + f]);
        L.zacc[f] += mx;
    } else {
        float sm = 0.0f;
        for (int j = 0; j < k; ++j) sm += h[j * HD + f];
        L.zacc[HD + f] += sm / (float)k;
    }
    __syncthreads();
}

// In-LDS GEMM: hout[n x 128] = hin[n x 128] @ B (pre-packed f16, 4 K-chunks)
__device__ inline void lds_gemm(const float* __restrict__ hin, int n,
                                const _Float16* __restrict__ Bp,
                                float* __restrict__ hout, int tid)
{
    const int wave = tid >> 5, lane = tid & 31;
    const int half = lane >> 4, l16 = lane & 15;
    const int mtiles = (n + 15) >> 4;
    const int ntiles = mtiles * 8;
    for (int t = wave; t < ntiles; t += 8) {
        const int row0 = (t >> 3) << 4;
        const int tn   = t & 7;
        v8f acc = {};
#pragma unroll
        for (int kk = 0; kk < 4; ++kk) {
            const int klo = (kk << 5) + half * 8;
            const int ar  = row0 + l16;
            const bool ok = (ar < n);
            v16h a;
            // stride 128 floats -> 32B aligned: float4 LDS loads
            const float4* rp = (const float4*)(hin + ar * HD);
            float4 q0, q1, q2, q3;
            if (ok) {
                q0 = rp[(klo >> 2) + 0];
                q1 = rp[(klo >> 2) + 1];
                q2 = rp[(klo >> 2) + 4];   // klo+16
                q3 = rp[(klo >> 2) + 5];
            } else {
                q0 = q1 = q2 = q3 = make_float4(0.f, 0.f, 0.f, 0.f);
            }
            a[0] = (_Float16)q0.x;  a[1] = (_Float16)q0.y;
            a[2] = (_Float16)q0.z;  a[3] = (_Float16)q0.w;
            a[4] = (_Float16)q1.x;  a[5] = (_Float16)q1.y;
            a[6] = (_Float16)q1.z;  a[7] = (_Float16)q1.w;
            a[8] = (_Float16)q2.x;  a[9] = (_Float16)q2.y;
            a[10] = (_Float16)q2.z; a[11] = (_Float16)q2.w;
            a[12] = (_Float16)q3.x; a[13] = (_Float16)q3.y;
            a[14] = (_Float16)q3.z; a[15] = (_Float16)q3.w;
            const v16h b = *(const v16h*)(Bp + (((tn * 4 + kk) * 32 + lane) << 4));
            acc = __builtin_amdgcn_wmma_f32_16x16x32_f16(false, a, false, b,
                                                         (short)0, acc, false, false);
        }
        const int crow = row0 + half * 8;
        const int ccol = tn * 16 + l16;
#pragma unroll
        for (int v = 0; v < 8; ++v)
            hout[(crow + v) * HD + ccol] = acc[v];
    }
}

// ===================== Kernel 2: whole graph pipeline per block ============
__global__ __launch_bounds__(256) void graph_kernel(
    const float* __restrict__ Hlin,
    const int* __restrict__ gsrc, const int* __restrict__ gdst,
    const float* __restrict__ b1, const _Float16* __restrict__ Bp2,
    const float* __restrict__ b2, const _Float16* __restrict__ Bp3,
    const float* __restrict__ b3,
    const float* __restrict__ p1, const float* __restrict__ p2,
    const float* __restrict__ p3,
    float* __restrict__ zout)
{
    extern __shared__ __align__(16) char raw[];
    GLds& L = *reinterpret_cast<GLds*>(raw);
    const int g = blockIdx.x;
    const int tid = threadIdx.x;

    L.zacc[tid] = 0.0f;

    // --- TDM: DMA this graph's 64KB feature tile (16384 x f32) into LDS h0 ---
    if (tid < 32) {
        const unsigned long long gaddr =
            (unsigned long long)(const void*)(Hlin + (size_t)g * NPER * HD);
        u32x4 g0 = {};
        g0[0] = 1u;                                        // count=1, user mode
        g0[1] = 0u;                                        // lds_addr = 0 (h0)
        g0[2] = (unsigned)(gaddr & 0xffffffffu);           // global_addr[31:0]
        g0[3] = (unsigned)((gaddr >> 32) & 0x01ffffffu)    // global_addr[56:32]
              | (2u << 30);                                // type = 2 (image)
        i32x8 g1 = {};
        g1[0] = 0x00020000;          // data_size = 2 (4B); mask=0; no pad/iter
        g1[1] = (int)0x40000000u;    // tensor_dim0[15:0]=0x4000 in [31:16]
        g1[2] = (int)0x00010000u;    // tensor_dim0[31:16]=0; tensor_dim1=1
        g1[3] = (int)0x40000000u;    // tile_dim0 = 0x4000 (16384 elements)
        g1[4] = 0;                   // tile_dim1 = 0, tile_dim2 = 0 (1-D tile)
        g1[5] = 16384;               // tensor_dim0_stride[31:0]
        g1[6] = 0;
        g1[7] = 0;
        i32x4 gz = {};
#if __clang_major__ >= 23
        i32x8 gz8 = {};
        __builtin_amdgcn_tensor_load_to_lds(g0, g1, gz, gz, gz8, 0);
#else
        __builtin_amdgcn_tensor_load_to_lds(g0, g1, gz, gz, 0);
#endif
        __builtin_amdgcn_s_wait_tensorcnt(0);
    }
    for (int e = tid; e < EPER; e += 256) {
        L.esrc[e] = gsrc[g * EPER + e] - g * NPER;
        L.edst[e] = gdst[g * EPER + e] - g * NPER;
    }
    __syncthreads();   // publishes TDM-filled h0 + edge lists to all waves

    // Layer 1
    gcn_agg(L, L.h0, L.h1, b1, NPER, tid);       // h1 = lrelu(gcn1)
    topk_pool(L, L.h1, L.h0, p1, NPER, K1, tid); // h0 = gated top-103
    readout_add(L, L.h0, K1, tid);               // r1

    // Layer 2
    lds_gemm(L.h0, K1, Bp2, L.h1, tid);          // h1 = h0 @ W2
    __syncthreads();
    gcn_agg(L, L.h1, L.h0, b2, K1, tid);
    topk_pool(L, L.h0, L.h1, p2, K1, K2, tid);
    readout_add(L, L.h1, K2, tid);               // r2

    // Layer 3
    lds_gemm(L.h1, K2, Bp3, L.h0, tid);
    __syncthreads();
    gcn_agg(L, L.h0, L.h1, b3, K2, tid);
    topk_pool(L, L.h1, L.h0, p3, K2, K3, tid);
    readout_add(L, L.h0, K3, tid);               // r3

    __syncthreads();
    zout[g * (2 * HD) + tid] = L.zacc[tid];
}

// ================ Kernel 3: T1 = bn1(lrelu(Z @ lin1_W + b)) ================
__global__ __launch_bounds__(256) void mlp1_kernel(
    const float* __restrict__ Z, const _Float16* __restrict__ BpL1,
    const float* __restrict__ bias,
    const float* __restrict__ bg, const float* __restrict__ bb,
    const float* __restrict__ brm, const float* __restrict__ brv,
    float* __restrict__ T1)
{
    const int tid = threadIdx.x;
    const int wave = tid >> 5, lane = tid & 31;
    const int half = lane >> 4, l16 = lane & 15;
    const int row0 = blockIdx.x * 16;                 // M = 512
    const int nt   = blockIdx.y * 8 + wave;           // N tile (of 64)
    v8f acc = {};
    const float4* zr = (const float4*)(Z + (row0 + l16) * 256); // 1KB-aligned rows
#pragma unroll
    for (int kk = 0; kk < 8; ++kk) {                  // K = 256
        const int klo = kk * 32 + half * 8;
        v16h a;
        float4 q0 = zr[(klo >> 2) + 0];
        float4 q1 = zr[(klo >> 2) + 1];
        float4 q2 = zr[(klo >> 2) + 4];
        float4 q3 = zr[(klo >> 2) + 5];
        a[0] = (_Float16)q0.x;  a[1] = (_Float16)q0.y;
        a[2] = (_Float16)q0.z;  a[3] = (_Float16)q0.w;
        a[4] = (_Float16)q1.x;  a[5] = (_Float16)q1.y;
        a[6] = (_Float16)q1.z;  a[7] = (_Float16)q1.w;
        a[8] = (_Float16)q2.x;  a[9] = (_Float16)q2.y;
        a[10] = (_Float16)q2.z; a[11] = (_Float16)q2.w;
        a[12] = (_Float16)q3.x; a[13] = (_Float16)q3.y;
        a[14] = (_Float16)q3.z; a[15] = (_Float16)q3.w;
        const v16h b = *(const v16h*)(BpL1 + (((nt * 8 + kk) * 32 + lane) << 4));
        acc = __builtin_amdgcn_wmma_f32_16x16x32_f16(false, a, false, b,
                                                     (short)0, acc, false, false);
    }
    const int crow = row0 + half * 8;
    const int ccol = nt * 16 + l16;
    const float gg = bg[ccol], be2 = bb[ccol];
    const float inv = rsqrtf(brv[ccol] + EPSBN);
    const float mm = brm[ccol], bs = bias[ccol];
#pragma unroll
    for (int v = 0; v < 8; ++v) {
        float x = acc[v] + bs;
        x = x > 0.0f ? x : NEG * x;
        x = (x - mm) * inv * gg + be2;
        T1[(crow + v) * 1024 + ccol] = x;
    }
}

// =========== Kernel 4: out = bn2(lrelu(T1@lin2_W+b2)) @ lin3_W + b3 ========
__global__ __launch_bounds__(64) void mlp2_kernel(
    const float* __restrict__ T1, const float* __restrict__ W2,
    const float* __restrict__ b2,
    const float* __restrict__ bg, const float* __restrict__ bb,
    const float* __restrict__ brm, const float* __restrict__ brv,
    const float* __restrict__ W3, const float* __restrict__ b3,
    float* __restrict__ out)
{
    __shared__ float s[64];
    const int g = blockIdx.x, t = threadIdx.x;    // 64 threads
    float acc = 0.0f;
    for (int k = 0; k < 1024; ++k)
        acc += T1[g * 1024 + k] * W2[k * 64 + t];
    acc += b2[t];
    acc = acc > 0.0f ? acc : NEG * acc;
    acc = (acc - brm[t]) * rsqrtf(brv[t] + EPSBN) * bg[t] + bb[t];
    s[t] = acc;
    __syncthreads();
    if (t < 2) {
        float o = b3[t];
        for (int j = 0; j < 64; ++j) o += s[j] * W3[j * 2 + t];
        out[g * 2 + t] = o;
    }
}

// ============================== Launcher ===================================
extern "C" void kernel_launch(void* const* d_in, const int* in_sizes, int n_in,
                              void* d_out, int out_size, void* d_ws, size_t ws_size,
                              hipStream_t stream)
{
    const float* x      = (const float*)d_in[0];
    const int*   src    = (const int*)d_in[1];
    const int*   dst    = (const int*)d_in[2];
    const float* W1     = (const float*)d_in[4];
    const float* b1     = (const float*)d_in[5];
    const float* W2     = (const float*)d_in[6];
    const float* b2     = (const float*)d_in[7];
    const float* W3     = (const float*)d_in[8];
    const float* b3     = (const float*)d_in[9];
    const float* p1     = (const float*)d_in[10];
    const float* p2     = (const float*)d_in[11];
    const float* p3     = (const float*)d_in[12];
    const float* lin1_W = (const float*)d_in[13];
    const float* lin1_b = (const float*)d_in[14];
    const float* lin2_W = (const float*)d_in[15];
    const float* lin2_b = (const float*)d_in[16];
    const float* lin3_W = (const float*)d_in[17];
    const float* lin3_b = (const float*)d_in[18];
    const float* bn1_g  = (const float*)d_in[19];
    const float* bn1_b  = (const float*)d_in[20];
    const float* bn1_rm = (const float*)d_in[21];
    const float* bn1_rv = (const float*)d_in[22];
    const float* bn2_g  = (const float*)d_in[23];
    const float* bn2_b  = (const float*)d_in[24];
    const float* bn2_rm = (const float*)d_in[25];
    const float* bn2_rv = (const float*)d_in[26];

    char* ws = (char*)d_ws;
    size_t off = 0;
    float* Hlin = (float*)(ws + off); off += (size_t)65536 * 128 * 4;   // 32 MB
    float* Zbuf = (float*)(ws + off); off += (size_t)512 * 256 * 4;
    float* T1   = (float*)(ws + off); off += (size_t)512 * 1024 * 4;
    _Float16* Bp1  = (_Float16*)(ws + off); off += (size_t)20480 * 2;   // 8nt*5kk*512
    _Float16* Bp2  = (_Float16*)(ws + off); off += (size_t)16384 * 2;   // 8nt*4kk*512
    _Float16* Bp3  = (_Float16*)(ws + off); off += (size_t)16384 * 2;
    _Float16* BpL1 = (_Float16*)(ws + off); off += (size_t)262144 * 2;  // 64nt*8kk*512
    float* outp = (float*)d_out;

    (void)hipFuncSetAttribute((const void*)graph_kernel,
                              hipFuncAttributeMaxDynamicSharedMemorySize,
                              (int)sizeof(GLds));

    // Pack weights into WMMA-B f16 fragment layout (tiny, runs once per call)
    pack_b_kernel<<<80,   256, 0, stream>>>(W1,     Bp1,  F_IN, 128,  5);
    pack_b_kernel<<<64,   256, 0, stream>>>(W2,     Bp2,  128,  128,  4);
    pack_b_kernel<<<64,   256, 0, stream>>>(W3,     Bp3,  128,  128,  4);
    pack_b_kernel<<<1024, 256, 0, stream>>>(lin1_W, BpL1, 256,  1024, 8);

    gemm1_kernel<<<512, 256, 0, stream>>>(x, Bp1, Hlin);
    graph_kernel<<<512, 256, sizeof(GLds), stream>>>(
        Hlin, src, dst, b1, Bp2, b2, Bp3, b3, p1, p2, p3, Zbuf);
    mlp1_kernel<<<dim3(32, 8), 256, 0, stream>>>(
        Zbuf, BpL1, lin1_b, bn1_g, bn1_b, bn1_rm, bn1_rv, T1);
    mlp2_kernel<<<512, 64, 0, stream>>>(
        T1, lin2_W, lin2_b, bn2_g, bn2_b, bn2_rm, bn2_rv, lin3_W, lin3_b, outp);
}
